// local_encoder_24369644437630
// MI455X (gfx1250) — matrix operand order repaired and checked by reading, not compiled, not run on hardware
//
#include <hip/hip_runtime.h>
#include <hip/hip_bf16.h>

typedef __attribute__((ext_vector_type(16))) _Float16 v16h;
typedef __attribute__((ext_vector_type(8)))  float    v8f;

#define B_SZ 32
#define N1   2048
#define NP   512
#define OUTC 832

static __device__ __forceinline__ float leaky(float h) { return h >= 0.f ? h : 0.2f * h; }

// ------------------------------------------------------------------ zero
__global__ void k_zero(float* p, int n) {
  int t = blockIdx.x * blockDim.x + threadIdx.x;
  if (t < n) p[t] = 0.f;
}

// ------------------------------------------------------------------ pack fp32 weights into WMMA f16 A-fragment layout
// layout: frag f = tile*(KPAD/32)+ks32 ; per frag: 32 lanes x 16 halfs (contiguous per lane)
template<int ROWS, int KREAL, int KPAD>
__global__ void k_pack(const float* __restrict__ w, _Float16* __restrict__ o) {
  int t = blockIdx.x * 256 + threadIdx.x;          // total = ROWS*KPAD halfs
  int e = t & 15;
  int l = (t >> 4) & 31;
  int f = t >> 9;
  int ks32 = f % (KPAD / 32);
  int tile = f / (KPAD / 32);
  int m  = tile * 16 + (l & 15);
  int kb = ks32 * 32 + ((l < 16) ? 0 : 16);
  int k  = kb + e;
  float v = (k < KREAL) ? w[(long)m * KREAL + k] : 0.f;
  o[t] = (_Float16)v;
}

// ------------------------------------------------------------------ kNN (top-k of -d^2, ties -> lower index)
template<int C, int M, int KNN>
__global__ void k_knn(const float* __restrict__ y, long ybs, int ycs, int* __restrict__ idxo) {
  int t = blockIdx.x * blockDim.x + threadIdx.x;
  int b = t / M, n = t % M;
  const float* Y = y + (long)b * ybs;
  float vals[KNN]; int ids[KNN];
#pragma unroll
  for (int j = 0; j < KNN; ++j) { vals[j] = -3.0e38f; ids[j] = 0; }
  constexpr int CC = (C <= 64) ? C : 1;
  float ctr[CC];
  if (C <= 64) {
#pragma unroll
    for (int c = 0; c < CC; ++c) ctr[c] = Y[c * ycs + n];
  }
  for (int m = 0; m < M; ++m) {
    float d2 = 0.f;
    if (C <= 64) {
#pragma unroll
      for (int c = 0; c < CC; ++c) { float df = ctr[c] - Y[c * ycs + m]; d2 = fmaf(df, df, d2); }
    } else {
      for (int c = 0; c < C; ++c) { float df = Y[c * ycs + n] - Y[c * ycs + m]; d2 = fmaf(df, df, d2); }
    }
    float v = -d2;
    if (v > vals[KNN - 1]) {
      vals[KNN - 1] = v; ids[KNN - 1] = m;
#pragma unroll
      for (int j = KNN - 1; j > 0; --j) {
        if (vals[j] > vals[j - 1]) {
          float tv = vals[j]; vals[j] = vals[j - 1]; vals[j - 1] = tv;
          int   ti = ids[j];  ids[j]  = ids[j - 1];  ids[j - 1] = ti;
        } else break;
      }
    }
  }
#pragma unroll
  for (int j = 0; j < KNN; ++j) idxo[(long)t * KNN + j] = ids[j];
}

// ------------------------------------------------------------------ BN statistics pass (sum / sumsq of h = Wa*feat + ba)
template<int CIN, int CMID, int KNN, int NPTS>
__global__ void k_stats(const float* __restrict__ y, long ybs, int ycs,
                        const int* __restrict__ idx,
                        const float* __restrict__ wa, const float* __restrict__ ba,
                        float* __restrict__ sums) {  // [CMID] sum, [CMID] sumsq
  constexpr int KR = 2 * CIN;
  constexpr int G  = 8;
  __shared__ float featS[G * KR];
  int tid = threadIdx.x;
  long pair0 = (long)blockIdx.x * G;
  for (int e = tid; e < G * KR; e += blockDim.x) {
    int g = e / KR, c = e % KR;
    long pair = pair0 + g;
    int b   = (int)(pair / ((long)NPTS * KNN));
    int rem = (int)(pair % ((long)NPTS * KNN));
    int n = rem / KNN, kk = rem % KNN;
    int cc = (c < CIN) ? c : c - CIN;
    const float* Yb = y + (long)b * ybs;
    float ctr = Yb[cc * ycs + n];
    float v;
    if (c < CIN) {
      int m = idx[((long)b * NPTS + n) * KNN + kk];
      v = ctr - Yb[cc * ycs + m];
    } else v = ctr;
    featS[e] = v;
  }
  __syncthreads();
  for (int c = tid; c < CMID; c += blockDim.x) {
    const float* wr = wa + (long)c * KR;
    float s = 0.f, s2 = 0.f;
    for (int g = 0; g < G; ++g) {
      float h = ba[c];
      for (int j = 0; j < KR; ++j) h = fmaf(wr[j], featS[g * KR + j], h);
      s += h; s2 = fmaf(h, h, s2);
    }
    atomicAdd(&sums[c], s);
    atomicAdd(&sums[CMID + c], s2);
  }
}

template<int CMID>
__global__ void k_finalize(const float* __restrict__ sums, const float* __restrict__ g,
                           const float* __restrict__ be, float cnt, float* __restrict__ aff) {
  int c = blockIdx.x * blockDim.x + threadIdx.x;
  if (c >= CMID) return;
  float mu  = sums[c] / cnt;
  float var = sums[CMID + c] / cnt - mu * mu;
  float a   = g[c] * rsqrtf(var + 1e-5f);
  aff[c]        = a;
  aff[CMID + c] = be[c] - mu * a;
}

// ------------------------------------------------------------------ fused edge-conv (WMMA), CG column-groups per workgroup
template<int CIN, int CMID, int COUT, int KNN, int NPTS, int CG>
__global__ void k_edge(const float* __restrict__ y, long ybs, int ycs,
                       const int* __restrict__ idx,
                       const _Float16* __restrict__ wpa, const float* __restrict__ ba,
                       const float* __restrict__ aff,
                       const _Float16* __restrict__ wpb, const float* __restrict__ bb,
                       float* __restrict__ out, long obs, int ocs) {
  constexpr int KR   = 2 * CIN;
  constexpr int KPAD = ((KR + 31) / 32) * 32;
  constexpr int TP   = 16 / KNN;                 // points per 16-column tile
  constexpr int PPW  = TP * CG;                  // points per workgroup
  __shared__ alignas(32) _Float16 featS[CG * 16 * KPAD];
  __shared__ alignas(32) _Float16 ahS[CG * 16 * CMID];
  int tid  = threadIdx.x;
  int lane = tid & 31, wv = tid >> 5;
  int gpb  = NPTS / PPW;
  int b    = blockIdx.x / gpb;
  int n0   = (blockIdx.x % gpb) * PPW;
  const float* Yb = y + (long)b * ybs;

  // Build f16 edge features: group cg, column = p*KNN + kk, row = channel (zero padded)
  for (int e = tid; e < CG * 16 * KPAD; e += 128) {
    int cg  = e / (16 * KPAD);
    int rem = e % (16 * KPAD);
    int col = rem / KPAD, c = rem % KPAD;
    float v = 0.f;
    if (c < KR) {
      int p = col / KNN, kk = col % KNN;
      int n = n0 + cg * TP + p;
      int cc = (c < CIN) ? c : c - CIN;
      float ctr = Yb[cc * ycs + n];
      if (c < CIN) {
        int m = idx[((long)b * NPTS + n) * KNN + kk];
        v = ctr - Yb[cc * ycs + m];
      } else v = ctr;
    }
    featS[e] = (_Float16)v;
  }
  __syncthreads();

  int col = lane & 15;
  // GEMM1 tiles (CMID/16) split across 4 waves; A fragment reused over CG groups
  for (int t = wv; t < CMID / 16; t += 4) {
    v8f acc[CG];
#pragma unroll
    for (int cg = 0; cg < CG; ++cg)
#pragma unroll
      for (int r = 0; r < 8; ++r) acc[cg][r] = 0.f;
    for (int ks32 = 0; ks32 < KPAD / 32; ++ks32) {
      v16h A = *(const v16h*)(wpa + ((long)(t * (KPAD / 32) + ks32) * 32 + lane) * 16);
      int klo = ks32 * 32 + ((lane < 16) ? 0 : 16);
#pragma unroll
      for (int cg = 0; cg < CG; ++cg) {
        v16h Bf = *(const v16h*)&featS[(cg * 16 + col) * KPAD + klo];
        acc[cg] = __builtin_amdgcn_wmma_f32_16x16x32_f16(false, A, false, Bf, (short)0, acc[cg], false, false);
      }
    }
#pragma unroll
    for (int cg = 0; cg < CG; ++cg) {
#pragma unroll
      for (int r = 0; r < 8; ++r) {
        int c = t * 16 + r + ((lane < 16) ? 0 : 8);
        float h = acc[cg][r] + ba[c];
        h = fmaf(h, aff[c], aff[CMID + c]);
        h = leaky(h);
        ahS[(cg * 16 + col) * CMID + c] = (_Float16)h;
      }
    }
  }
  __syncthreads();

  // GEMM2 tiles (COUT/16), then max over the KNN columns of each point
  for (int t = wv; t < COUT / 16; t += 4) {
    v8f acc[CG];
#pragma unroll
    for (int cg = 0; cg < CG; ++cg)
#pragma unroll
      for (int r = 0; r < 8; ++r) acc[cg][r] = 0.f;
    for (int ks32 = 0; ks32 < CMID / 32; ++ks32) {
      v16h A = *(const v16h*)(wpb + ((long)(t * (CMID / 32) + ks32) * 32 + lane) * 16);
      int klo = ks32 * 32 + ((lane < 16) ? 0 : 16);
#pragma unroll
      for (int cg = 0; cg < CG; ++cg) {
        v16h Bf = *(const v16h*)&ahS[(cg * 16 + col) * CMID + klo];
        acc[cg] = __builtin_amdgcn_wmma_f32_16x16x32_f16(false, A, false, Bf, (short)0, acc[cg], false, false);
      }
    }
#pragma unroll
    for (int cg = 0; cg < CG; ++cg) {
#pragma unroll
      for (int r = 0; r < 8; ++r) {
        float v = acc[cg][r];
        if (KNN > 1) v = fmaxf(v, __shfl_xor(v, 1, 32));
        if (KNN > 2) v = fmaxf(v, __shfl_xor(v, 2, 32));
        if (KNN > 4) v = fmaxf(v, __shfl_xor(v, 4, 32));
        if (KNN > 8) v = fmaxf(v, __shfl_xor(v, 8, 32));
        if ((col % KNN) == 0) {
          int p = col / KNN;
          int o = t * 16 + r + ((lane < 16) ? 0 : 8);
          out[(long)b * obs + (long)o * ocs + (n0 + cg * TP + p)] = v + bb[o];
        }
      }
    }
  }
}

// ------------------------------------------------------------------ farthest point sampling (serial 512 steps, 1 WG per batch)
__global__ void k_fps(const float* __restrict__ x, int* __restrict__ fpsidx) {
  __shared__ float dists[N1];
  __shared__ float rv[256];
  __shared__ int   ri[256];
  int b = blockIdx.x, tid = threadIdx.x;
  const float* X = x + (long)b * 3 * N1;
  for (int j = tid; j < N1; j += 256) dists[j] = 1e10f;
  if (tid == 0) fpsidx[b * NP] = 0;
  int last = 0;
  __syncthreads();
  for (int i = 1; i < NP; ++i) {
    float px = X[last], py = X[N1 + last], pz = X[2 * N1 + last];
    float bv = -1.f; int bi = 0;
    for (int j = tid; j < N1; j += 256) {
      float dx = X[j] - px, dy = X[N1 + j] - py, dz = X[2 * N1 + j] - pz;
      float d  = dx * dx + dy * dy + dz * dz;
      float nd = fminf(dists[j], d);
      dists[j] = nd;
      if (nd > bv) { bv = nd; bi = j; }
    }
    rv[tid] = bv; ri[tid] = bi;
    __syncthreads();
    for (int s = 128; s > 0; s >>= 1) {
      if (tid < s) {
        float ov = rv[tid + s]; int oi = ri[tid + s];
        if (ov > rv[tid] || (ov == rv[tid] && oi < ri[tid])) { rv[tid] = ov; ri[tid] = oi; }
      }
      __syncthreads();
    }
    last = ri[0];
    if (tid == 0) fpsidx[b * NP + i] = last;
    __syncthreads();
  }
}

__global__ void k_gather(const float* __restrict__ x1, const int* __restrict__ fpsidx,
                         float* __restrict__ out) {
  int t = blockIdx.x * 256 + threadIdx.x;          // over B*64*NP
  int b = t / (64 * NP); int r = t % (64 * NP);
  int c = r / NP, j = r % NP;
  out[(long)b * OUTC * NP + c * NP + j] = x1[((long)b * 64 + c) * N1 + fpsidx[b * NP + j]];
}

// ------------------------------------------------------------------ launch
extern "C" void kernel_launch(void* const* d_in, const int* in_sizes, int n_in,
                              void* d_out, int out_size, void* d_ws, size_t ws_size,
                              hipStream_t stream) {
  (void)in_sizes; (void)n_in; (void)out_size; (void)ws_size;
  const float* x   = (const float*)d_in[0];
  const float* w1a = (const float*)d_in[1];  const float* b1a = (const float*)d_in[2];
  const float* g1  = (const float*)d_in[3];  const float* be1 = (const float*)d_in[4];
  const float* w1b = (const float*)d_in[5];  const float* b1b = (const float*)d_in[6];
  const float* w2a = (const float*)d_in[7];  const float* b2a = (const float*)d_in[8];
  const float* g2  = (const float*)d_in[9];  const float* be2 = (const float*)d_in[10];
  const float* w2b = (const float*)d_in[11]; const float* b2b = (const float*)d_in[12];
  const float* w3a = (const float*)d_in[13]; const float* b3a = (const float*)d_in[14];
  const float* g3  = (const float*)d_in[15]; const float* be3 = (const float*)d_in[16];
  const float* w3b = (const float*)d_in[17]; const float* b3b = (const float*)d_in[18];
  float* out = (float*)d_out;

  char* ws = (char*)d_ws; size_t off = 0;
  auto carve = [&](size_t bytes) -> void* {
    void* p = ws + off;
    off = (off + bytes + 255) & ~(size_t)255;
    return p;
  };
  int*   idx1  = (int*)  carve((size_t)B_SZ * N1 * 16 * sizeof(int));
  int*   idx2  = (int*)  carve((size_t)B_SZ * NP * 8  * sizeof(int));
  int*   idx3  = (int*)  carve((size_t)B_SZ * NP * 4  * sizeof(int));
  int*   fpsi  = (int*)  carve((size_t)B_SZ * NP      * sizeof(int));
  float* x1    = (float*)carve((size_t)B_SZ * 64 * N1 * sizeof(float));
  float* stats = (float*)carve(832 * sizeof(float));   // sums: L1[64] L2[256] L3[512]
  float* affb  = (float*)carve(832 * sizeof(float));   // affine A,B per layer
  _Float16* p1a = (_Float16*)carve(  1024 * sizeof(_Float16));  // 32x32
  _Float16* p1b = (_Float16*)carve(  2048 * sizeof(_Float16));  // 64x32
  _Float16* p2a = (_Float16*)carve( 16384 * sizeof(_Float16));  // 128x128
  _Float16* p2b = (_Float16*)carve( 32768 * sizeof(_Float16));  // 256x128
  _Float16* p3a = (_Float16*)carve(131072 * sizeof(_Float16));  // 256x512
  _Float16* p3b = (_Float16*)carve(131072 * sizeof(_Float16));  // 512x256
  float* sums1 = stats;       float* sums2 = stats + 64;  float* sums3 = stats + 320;
  float* aff1  = affb;        float* aff2  = affb + 64;   float* aff3  = affb + 320;

  k_zero<<<4, 256, 0, stream>>>(stats, 832);
  k_pack< 32,   6,  32><<<   4, 256, 0, stream>>>(w1a, p1a);
  k_pack< 64,  32,  32><<<   8, 256, 0, stream>>>(w1b, p1b);
  k_pack<128, 128, 128><<<  64, 256, 0, stream>>>(w2a, p2a);
  k_pack<256, 128, 128><<< 128, 256, 0, stream>>>(w2b, p2b);
  k_pack<256, 512, 512><<< 512, 256, 0, stream>>>(w3a, p3a);
  k_pack<512, 256, 256><<< 512, 256, 0, stream>>>(w3b, p3b);

  // ---- layer 1 (Cin=3 -> 32 -> 64, k=16, N=2048) ----
  k_knn<3, N1, 16><<<B_SZ * N1 / 256, 256, 0, stream>>>(x, (long)3 * N1, N1, idx1);
  k_stats<3, 32, 16, N1><<<B_SZ * N1 * 16 / 8, 128, 0, stream>>>(x, (long)3 * N1, N1, idx1, w1a, b1a, sums1);
  k_finalize<32><<<1, 32, 0, stream>>>(sums1, g1, be1, (float)((long)B_SZ * N1 * 16), aff1);
  k_edge<3, 32, 64, 16, N1, 4><<<B_SZ * (N1 / 4), 128, 0, stream>>>(x, (long)3 * N1, N1, idx1,
      p1a, b1a, aff1, p1b, b1b, x1, (long)64 * N1, N1);

  // ---- FPS + gather into d_out channels [0,64) ----
  k_fps<<<B_SZ, 256, 0, stream>>>(x, fpsi);
  k_gather<<<B_SZ * 64 * NP / 256, 256, 0, stream>>>(x1, fpsi, out);

  const long ybs = (long)OUTC * NP; const int ycs = NP;

  // ---- layer 2 (64 -> 128 -> 256, k=8, N=512), writes d_out channels [64,320) ----
  const float* y2 = out;
  k_knn<64, NP, 8><<<B_SZ * NP / 256, 256, 0, stream>>>(y2, ybs, ycs, idx2);
  k_stats<64, 128, 8, NP><<<B_SZ * NP * 8 / 8, 128, 0, stream>>>(y2, ybs, ycs, idx2, w2a, b2a, sums2);
  k_finalize<128><<<1, 128, 0, stream>>>(sums2, g2, be2, (float)((long)B_SZ * NP * 8), aff2);
  k_edge<64, 128, 256, 8, NP, 4><<<B_SZ * (NP / 8), 128, 0, stream>>>(y2, ybs, ycs, idx2,
      p2a, b2a, aff2, p2b, b2b, out + 64 * NP, ybs, NP);

  // ---- layer 3 (256 -> 256 -> 512, k=4, N=512), writes d_out channels [320,832) ----
  const float* y3 = out + 64 * NP;
  k_knn<256, NP, 4><<<B_SZ * NP / 256, 256, 0, stream>>>(y3, ybs, ycs, idx3);
  k_stats<256, 256, 4, NP><<<B_SZ * NP * 4 / 8, 128, 0, stream>>>(y3, ybs, ycs, idx3, w3a, b3a, sums3);
  k_finalize<256><<<1, 256, 0, stream>>>(sums3, g3, be3, (float)((long)B_SZ * NP * 4), aff3);
  k_edge<256, 256, 512, 4, NP, 4><<<B_SZ * (NP / 16), 128, 0, stream>>>(y3, ybs, ycs, idx3,
      p3a, b3a, aff3, p3b, b3b, out + 320 * NP, ybs, NP);
}
